// Pool_24481313587489
// MI455X (gfx1250) — 8-bit matrix operand order repaired and checked by reading, compiled, not run on hardware
//
#include <hip/hip_runtime.h>
#include <hip/hip_bf16.h>

typedef int v8i __attribute__((ext_vector_type(8)));
typedef unsigned int u32x4 __attribute__((ext_vector_type(4)));
typedef unsigned int u32x8 __attribute__((ext_vector_type(8)));

#define N_NODES 8192
#define D_FEAT  512
#define KK      4096
#define KTILES  128
#define MTILES  256
#define NTILES  256

__global__ __launch_bounds__(256)
void scores_kernel(const float* __restrict__ ht, const float* __restrict__ W,
                   const float* __restrict__ b, float* __restrict__ scores) {
    const int wave = threadIdx.x >> 5;
    const int lane = threadIdx.x & 31;
    const int row  = blockIdx.x * 8 + wave;
    const float* hrow = ht + (size_t)row * D_FEAT;
    float sum = 0.f;
    #pragma unroll
    for (int j = 0; j < D_FEAT / 32; ++j) {
        int c = lane + j * 32;
        sum += hrow[c] * W[c];
    }
    #pragma unroll
    for (int off = 16; off > 0; off >>= 1)
        sum += __shfl_down(sum, off, 32);
    if (lane == 0) {
        float z = sum + b[0];
        scores[row] = 1.0f / (1.0f + __expf(-z));
    }
}

__global__ __launch_bounds__(1024)
void topk_sort_kernel(const float* __restrict__ scores,
                      int* __restrict__ topk_idx, float* __restrict__ topk_val) {
    extern __shared__ unsigned long long keys[];
    const int n = N_NODES;
    for (int i = threadIdx.x; i < n; i += 1024) {
        unsigned int ub = __float_as_uint(scores[i]);
        keys[i] = ((unsigned long long)(~ub) << 32) | (unsigned int)i;
    }
    __syncthreads();
    for (int k = 2; k <= n; k <<= 1) {
        for (int j = k >> 1; j > 0; j >>= 1) {
            for (int i = threadIdx.x; i < n; i += 1024) {
                int ixj = i ^ j;
                if (ixj > i) {
                    unsigned long long a = keys[i], c = keys[ixj];
                    bool up = ((i & k) == 0);
                    if (up ? (a > c) : (a < c)) { keys[i] = c; keys[ixj] = a; }
                }
            }
            __syncthreads();
        }
    }
    for (int i = threadIdx.x; i < KK; i += 1024) {
        unsigned long long kkey = keys[i];
        topk_idx[i] = (int)(unsigned int)(kkey & 0xFFFFFFFFull);
        topk_val[i] = __uint_as_float(~(unsigned int)(kkey >> 32));
    }
}

__global__ __launch_bounds__(256)
void packA_kernel(const int* __restrict__ g, const int* __restrict__ topk_idx,
                  unsigned int* __restrict__ Apack) {
    const int frag  = blockIdx.x;
    const int ktile = frag & (KTILES - 1);
    const int mtile = frag >> 7;
    const int lane  = threadIdx.x >> 3;
    const int v     = threadIdx.x & 7;
    const int mloc  = lane & 15;
    const int grp   = lane >> 4;
    const int klocal = ((v >> 2) << 5) + (grp << 4) + ((v & 3) << 2);
    const int row   = topk_idx[mtile * 16 + mloc];
    const int* gp   = g + (size_t)row * N_NODES + (ktile * 64 + klocal);
    unsigned int u = (unsigned)(gp[0] != 0)
                   | ((unsigned)(gp[1] != 0) << 8)
                   | ((unsigned)(gp[2] != 0) << 16)
                   | ((unsigned)(gp[3] != 0) << 24);
    Apack[(size_t)frag * 256 + threadIdx.x] = u;
}

__global__ __launch_bounds__(256)
void packB_kernel(const int* __restrict__ g, const int* __restrict__ topk_idx,
                  unsigned int* __restrict__ Bpack) {
    const int frag  = blockIdx.x;
    const int ntile = frag & (NTILES - 1);
    const int ktile = frag >> 8;
    const int lane  = threadIdx.x >> 3;
    const int v     = threadIdx.x & 7;
    const int nloc  = lane & 15;
    const int grp   = lane >> 4;
    const int klocal = ((v >> 2) << 5) + (grp << 4) + ((v & 3) << 2);
    const int col   = topk_idx[ntile * 16 + nloc];
    const int kbase = ktile * 64 + klocal;
    unsigned int u = 0;
    #pragma unroll
    for (int bb = 0; bb < 4; ++bb) {
        int gv = g[(size_t)(kbase + bb) * N_NODES + col];
        u |= ((unsigned)(gv != 0)) << (bb * 8);
    }
    Bpack[(size_t)frag * 256 + threadIdx.x] = u;
}

__device__ __forceinline__ void tdm_load_2d(unsigned lds_off, const void* gaddr,
                                            unsigned tile_d0_e, unsigned tile_d1,
                                            unsigned long long stride_e) {
    unsigned long long ga = (unsigned long long)(size_t)gaddr;
    u32x4 g0;
    g0[0] = 1u;
    g0[1] = lds_off;
    g0[2] = (unsigned)(ga & 0xFFFFFFFFu);
    g0[3] = (unsigned)((ga >> 32) & 0x1FFFFFFu) | (2u << 30);
    u32x8 g1;
    g1[0] = (3u << 16);
    g1[1] = (tile_d0_e & 0xFFFFu) << 16;
    g1[2] = ((tile_d0_e >> 16) & 0xFFFFu)
          | ((tile_d1 & 0xFFFFu) << 16);
    g1[3] = ((tile_d1 >> 16) & 0xFFFFu)
          | ((tile_d0_e & 0xFFFFu) << 16);
    g1[4] = (tile_d1 & 0xFFFFu);
    g1[5] = (unsigned)(stride_e & 0xFFFFFFFFu);
    g1[6] = (unsigned)((stride_e >> 32) & 0xFFFFu);
    g1[7] = 0u;
    asm volatile("tensor_load_to_lds %0, %1" : : "s"(g0), "s"(g1) : "memory");
}

__global__ __launch_bounds__(512)
void gemm_iu8_kernel(const unsigned char* __restrict__ Apack,
                     const unsigned char* __restrict__ Bpack,
                     float* __restrict__ outg) {
    extern __shared__ unsigned char smem[];
    const unsigned lds_base = __builtin_amdgcn_groupstaticsize();

    const int lane = threadIdx.x & 31;
    const int wave = threadIdx.x >> 5;
    const int wm = wave & 3;
    const int wn = wave >> 2;
    const int mtile_blk = blockIdx.x * 16;
    const int ntile_blk = blockIdx.y * 16;
    const int mtile0 = mtile_blk + wm * 4;
    const int ntile0 = ntile_blk + wn * 4;

    auto issue = [&](int kt, int buf) {
        tdm_load_2d(lds_base + (unsigned)buf * 32768u,
                    Apack + ((size_t)mtile_blk * KTILES + kt) * 1024,
                     128u,  16u,
                     (unsigned long long)(KTILES * 1024 / 8));
        tdm_load_2d(lds_base + (unsigned)buf * 32768u + 16384u,
                    Bpack + ((size_t)kt * NTILES + ntile_blk) * 1024,
                     2048u,  1u,
                     2048ull);
    };

    v8i acc[4][4] = {};

    if (wave == 0) {
        issue(0, 0);
        __builtin_amdgcn_s_wait_tensorcnt(0);
    }
    __syncthreads();

    for (int kt = 0; kt < KTILES; ++kt) {
        const int buf = kt & 1;
        const unsigned char* sA = smem + buf * 32768;
        const unsigned char* sB = sA + 16384;

        v8i afr[4], bfr[4];
        #pragma unroll
        for (int mi = 0; mi < 4; ++mi)
            afr[mi] = *(const v8i*)(sA + (size_t)(wm * 4 + mi) * 1024 + (size_t)lane * 32);
        #pragma unroll
        for (int ni = 0; ni < 4; ++ni)
            bfr[ni] = *(const v8i*)(sB + (size_t)(wn * 4 + ni) * 1024 + (size_t)lane * 32);

        if (wave == 0 && kt + 1 < KTILES)
            issue(kt + 1, buf ^ 1);

        #pragma unroll
        for (int mi = 0; mi < 4; ++mi)
            #pragma unroll
            for (int ni = 0; ni < 4; ++ni)
                acc[mi][ni] = __builtin_amdgcn_wmma_i32_16x16x64_iu8(
                    false, afr[mi], false, bfr[ni], acc[mi][ni], false, false);

        if (kt + 1 < KTILES) {
            if (wave == 0)
                __builtin_amdgcn_s_wait_tensorcnt(0);
            __syncthreads();
        }
    }

    const int nloc  = lane & 15;
    const int mhigh = (lane >> 4) * 8;
    #pragma unroll
    for (int mi = 0; mi < 4; ++mi) {
        const int rowbase = (mtile0 + mi) * 16 + mhigh;
        #pragma unroll
        for (int ni = 0; ni < 4; ++ni) {
            const int col = (ntile0 + ni) * 16 + nloc;
            #pragma unroll
            for (int r = 0; r < 8; ++r)
                outg[(size_t)(rowbase + r) * KK + col] = (acc[mi][ni][r] > 0) ? 1.0f : 0.0f;
        }
    }
}

__global__ __launch_bounds__(256)
void deg_kernel(const float* __restrict__ outg, float* __restrict__ deg) {
    __shared__ float red[256];
    const int row = blockIdx.x;
    const float* p = outg + (size_t)row * KK;
    float s = 0.f;
    for (int j = threadIdx.x; j < KK; j += 256) s += p[j];
    red[threadIdx.x] = s;
    __syncthreads();
    for (int w = 128; w > 0; w >>= 1) {
        if (threadIdx.x < w) red[threadIdx.x] += red[threadIdx.x + w];
        __syncthreads();
    }
    if (threadIdx.x == 0) deg[row] = red[0];
}

__global__ __launch_bounds__(256)
void scale_kernel(float* __restrict__ outg, const float* __restrict__ deg) {
    size_t t = (size_t)blockIdx.x * 256 + threadIdx.x;
    int col = (int)(t & (KK - 1));
    outg[t] = outg[t] / deg[col];
}

__global__ __launch_bounds__(256)
void gather_scale_kernel(const float* __restrict__ src, const int* __restrict__ topk_idx,
                         const float* __restrict__ topk_val, float* __restrict__ dst) {
    size_t t = (size_t)blockIdx.x * 256 + threadIdx.x;
    int i = (int)(t >> 7);
    int c = (int)(t & 127);
    float v = topk_val[i];
    const float4* s = (const float4*)(src + (size_t)topk_idx[i] * D_FEAT) + c;
    float4 x = *s;
    x.x *= v; x.y *= v; x.z *= v; x.w *= v;
    ((float4*)(dst + (size_t)i * D_FEAT))[c] = x;
}

extern "C" void kernel_launch(void* const* d_in, const int* in_sizes, int n_in,
                              void* d_out, int out_size, void* d_ws, size_t ws_size,
                              hipStream_t stream) {
    (void)in_sizes; (void)n_in; (void)out_size; (void)ws_size;
    const float* ht = (const float*)d_in[0];
    const float* hs = (const float*)d_in[1];
    const int*   g  = (const int*)  d_in[2];
    const float* W  = (const float*)d_in[3];
    const float* b  = (const float*)d_in[4];

    char* ws = (char*)d_ws;
    float* scores    = (float*)ws;
    int*   tidx      = (int*)  (ws + 32768);
    float* tval      = (float*)(ws + 49152);
    float* deg       = (float*)(ws + 65536);
    unsigned int* Apack = (unsigned int*)(ws + 131072);
    unsigned int* Bpack = (unsigned int*)(ws + 131072 + 33554432);

    float* out    = (float*)d_out;
    float* out_ht = out;
    float* out_hs = out + (size_t)KK * D_FEAT;
    float* out_g  = out + 2 * (size_t)KK * D_FEAT;

    scores_kernel<<<N_NODES / 8, 256, 0, stream>>>(ht, W, b, scores);
    topk_sort_kernel<<<1, 1024, N_NODES * sizeof(unsigned long long), stream>>>(scores, tidx, tval);
    packA_kernel<<<MTILES * KTILES, 256, 0, stream>>>(g, tidx, Apack);
    packB_kernel<<<KTILES * NTILES, 256, 0, stream>>>(g, tidx, Bpack);
    gemm_iu8_kernel<<<dim3(16, 16), 512, 65536, stream>>>(
        (const unsigned char*)Apack, (const unsigned char*)Bpack, out_g);
    deg_kernel<<<KK, 256, 0, stream>>>(out_g, deg);
    scale_kernel<<<(KK * KK) / 256, 256, 0, stream>>>(out_g, deg);
    gather_scale_kernel<<<(KK * D_FEAT / 4) / 256, 256, 0, stream>>>(ht, tidx, tval, out_ht);
    gather_scale_kernel<<<(KK * D_FEAT / 4) / 256, 256, 0, stream>>>(hs, tidx, tval, out_hs);
}
